// VQVAE_37134287241728
// MI455X (gfx1250) — matrix operand order repaired and checked by reading, not compiled, not run on hardware
//
#include <hip/hip_runtime.h>
#include <cstddef>

// ---------------------------------------------------------------------------
// Residual-VQ point-cloud autoencoder forward, MI455X (gfx1250, wave32).
// All dense math goes through V_WMMA_F32_16X16X32_F16 (f16 in, f32 acc).
// Every GEMM is LDS-free: fragments are built straight from global memory
// (float4 loads; clamped/select loads for ragged K or N), 32x16 tile per wave
// (2 WMMAs share one B fragment), K-loop unrolled x4.
// ---------------------------------------------------------------------------

#define BATCH 32
#define NPTS  2048
#define HID   256
#define KC    1024
#define RES   12
#define NVOX  1728     // 12^3
#define OSP   216      // 6^3 conv output cells
#define RECN  (BATCH * OSP * 3)   // 20736 floats of reconstruction

typedef __attribute__((ext_vector_type(16))) _Float16 v16h;
typedef __attribute__((ext_vector_type(8)))  float    v8f;

__device__ __forceinline__ v16h pack16(float4 a, float4 b, float4 c, float4 d)
{
    v16h r;
    r[0]  = (_Float16)a.x; r[1]  = (_Float16)a.y; r[2]  = (_Float16)a.z; r[3]  = (_Float16)a.w;
    r[4]  = (_Float16)b.x; r[5]  = (_Float16)b.y; r[6]  = (_Float16)b.z; r[7]  = (_Float16)b.w;
    r[8]  = (_Float16)c.x; r[9]  = (_Float16)c.y; r[10] = (_Float16)c.z; r[11] = (_Float16)c.w;
    r[12] = (_Float16)d.x; r[13] = (_Float16)d.y; r[14] = (_Float16)d.z; r[15] = (_Float16)d.w;
    return r;
}

// per-lane A row pointers for a 32-row tile (rows m0+col and m0+16+col)
__device__ __forceinline__ void a_row_ptrs(const float* __restrict__ A, int rowsUsed,
                                           int strideRows, int K, int m0, int col,
                                           const float*& arow0, const float*& arow1)
{
    if (rowsUsed == strideRows) {               // compact layout: no division at all
        arow0 = A + (size_t)(m0 + col) * (size_t)K;
        arow1 = A + (size_t)(m0 + 16 + col) * (size_t)K;
    } else {
        int r0 = m0 + col;      int b0 = r0 / rowsUsed;
        arow0 = A + ((size_t)b0 * strideRows + (r0 - b0 * rowsUsed)) * (size_t)K;
        int r1 = r0 + 16;       int b1 = r1 / rowsUsed;
        arow1 = A + ((size_t)b1 * strideRows + (r1 - b1 * rowsUsed)) * (size_t)K;
    }
}

// ---------------------------------------------------------------------------
// FAST WMMA GEMM (requires K%32==0, M%32==0, N%16==0).
//   A row m -> batch b = m / rowsUsed, row ptr A + (b*strideRows + m%rowsUsed)*K
//   BCOL=0 (row-major): B(k,n) = W[boff + k*ldbK + n]       (ldbN==1)
//   BCOL=1 (col-major): B(k,n) = W[boff + n*ldbN + k]       (ldbK==1, codebook^T)
// Fragment layouts used here (contiguous k order, same for A and B):
//   A lane(l): row = l&15, half j -> k = (l>>4)*16 + j.
//     => one contiguous 16-float row chunk -> four float4 loads.
//   B lane(l): col = l&15, half j -> k = (l>>4)*16 + j.
//     => col-major: one contiguous 16-float chunk; row-major: 16 coalesced dwords
//        via an incrementally-stepped per-lane pointer (no per-step multiplies).
// ---------------------------------------------------------------------------
template <int BCOL>
__global__ __launch_bounds__(256)
void k_gemm_fast(const float* __restrict__ A, int rowsUsed, int strideRows,
                 const float* __restrict__ W, long long ldbK, long long ldbN, long long boff,
                 const float* __restrict__ bias,
                 float* __restrict__ C,
                 int M, int K, int N, int tilesM, int tilesN,
                 int accC, int relu)
{
    const int wave = threadIdx.x >> 5, lane = threadIdx.x & 31;
    const int tile = blockIdx.x * 8 + wave;
    if (tile >= tilesM * tilesN) return;        // wave-uniform exit, EXEC stays full
    const int tm = tile % tilesM, tn = tile / tilesM;
    const int m0 = tm * 32, n0 = tn * 16;
    const int col = lane & 15, half = lane >> 4;

    const float *arow0, *arow1;
    a_row_ptrs(A, rowsUsed, strideRows, K, m0, col, arow0, arow1);

    // per-lane B pointers, stepped incrementally through the K loop
    const float* bcol = W + boff + (size_t)(n0 + col) * (size_t)(BCOL ? ldbN : 0);
    const float* bpr  = W + boff + (size_t)(half * 16) * (size_t)ldbK + (n0 + col);

    v8f acc0, acc1;
    if (accC) {
#pragma unroll
        for (int g = 0; g < 8; ++g) {
            acc0[g] = C[(size_t)(m0 + half * 8 + g) * N + (n0 + col)];
            acc1[g] = C[(size_t)(m0 + 16 + half * 8 + g) * N + (n0 + col)];
        }
    } else {
        float bv = bias ? bias[n0 + col] : 0.f;
#pragma unroll
        for (int g = 0; g < 8; ++g) { acc0[g] = bv; acc1[g] = bv; }
    }

#pragma unroll 4
    for (int k0 = 0; k0 < K; k0 += 32) {
        const float4* pa0 = (const float4*)(arow0 + k0 + half * 16);
        const float4* pa1 = (const float4*)(arow1 + k0 + half * 16);
        v16h af0 = pack16(pa0[0], pa0[1], pa0[2], pa0[3]);
        v16h af1 = pack16(pa1[0], pa1[1], pa1[2], pa1[3]);
        v16h bf;
        if (BCOL) {
            const float4* pb = (const float4*)(bcol + k0 + half * 16);
            bf = pack16(pb[0], pb[1], pb[2], pb[3]);
        } else {
#pragma unroll
            for (int j = 0; j < 16; ++j)
                bf[j] = (_Float16)bpr[(size_t)j * ldbK];
            bpr += 32 * ldbK;
        }
        acc0 = __builtin_amdgcn_wmma_f32_16x16x32_f16(
                   false, af0, false, bf, (short)0, acc0, false, false);
        acc1 = __builtin_amdgcn_wmma_f32_16x16x32_f16(
                   false, af1, false, bf, (short)0, acc1, false, false);
    }

#pragma unroll
    for (int g = 0; g < 8; ++g) {
        float v0 = acc0[g], v1 = acc1[g];
        if (relu) { v0 = fmaxf(v0, 0.f); v1 = fmaxf(v1, 0.f); }
        C[(size_t)(m0 + half * 8 + g) * N + (n0 + col)] = v0;
        C[(size_t)(m0 + 16 + half * 8 + g) * N + (n0 + col)] = v1;
    }
}

// ---------------------------------------------------------------------------
// EDGE WMMA GEMM: row-major B, M%32==0, but ragged K (tail step with clamped
// loads + selects) and/or ragged N (clamped B fetch, guarded stores).
// Branch-free fragment construction: clamp address into range, cndmask to 0.
// Used for the encoder input layer (K=3) and final projection (N=3).
// ---------------------------------------------------------------------------
__global__ __launch_bounds__(256)
void k_gemm_edge(const float* __restrict__ A, int rowsUsed, int strideRows,
                 const float* __restrict__ W, long long ldbK, long long boff,
                 const float* __restrict__ bias,
                 float* __restrict__ C,
                 int M, int K, int N, int tilesM, int tilesN, int relu)
{
    const int wave = threadIdx.x >> 5, lane = threadIdx.x & 31;
    const int tile = blockIdx.x * 8 + wave;
    if (tile >= tilesM * tilesN) return;
    const int tm = tile % tilesM, tn = tile / tilesM;
    const int m0 = tm * 32, n0 = tn * 16;
    const int col = lane & 15, half = lane >> 4;

    const float *arow0, *arow1;
    a_row_ptrs(A, rowsUsed, strideRows, K, m0, col, arow0, arow1);

    const int  n   = n0 + col;
    const bool nok = n < N;
    const int  nc  = nok ? n : N - 1;                  // clamped, always in range
    const float* bcolp = W + boff + nc;                // B(k,nc) = bcolp[k*ldbK]

    float bv = bias ? bias[nc] : 0.f;
    if (!nok) bv = 0.f;
    v8f acc0, acc1;
#pragma unroll
    for (int g = 0; g < 8; ++g) { acc0[g] = bv; acc1[g] = bv; }

    int k0 = 0;
    for (; k0 + 32 <= K; k0 += 32) {                   // full steps: float4 A path
        const float4* pa0 = (const float4*)(arow0 + k0 + half * 16);
        const float4* pa1 = (const float4*)(arow1 + k0 + half * 16);
        v16h af0 = pack16(pa0[0], pa0[1], pa0[2], pa0[3]);
        v16h af1 = pack16(pa1[0], pa1[1], pa1[2], pa1[3]);
        v16h bf;
#pragma unroll
        for (int j = 0; j < 16; ++j) {
            float v = bcolp[(size_t)(k0 + half * 16 + j) * ldbK];
            bf[j] = nok ? (_Float16)v : (_Float16)0.f;
        }
        acc0 = __builtin_amdgcn_wmma_f32_16x16x32_f16(
                   false, af0, false, bf, (short)0, acc0, false, false);
        acc1 = __builtin_amdgcn_wmma_f32_16x16x32_f16(
                   false, af1, false, bf, (short)0, acc1, false, false);
    }
    if (k0 < K) {                                      // ragged K tail (1..31)
        v16h af0, af1, bf;
#pragma unroll
        for (int j = 0; j < 16; ++j) {
            int ka = half * 16 + j;
            int kk = k0 + ka;
            int kcl = kk < K ? kk : 0;                 // clamped address, valid load
            float v0 = arow0[kcl], v1 = arow1[kcl];
            af0[j] = (kk < K) ? (_Float16)v0 : (_Float16)0.f;
            af1[j] = (kk < K) ? (_Float16)v1 : (_Float16)0.f;
            int kb = k0 + half * 16 + j;
            int kbc = kb < K ? kb : 0;
            float vb = bcolp[(size_t)kbc * ldbK];
            bf[j] = (nok && kb < K) ? (_Float16)vb : (_Float16)0.f;
        }
        acc0 = __builtin_amdgcn_wmma_f32_16x16x32_f16(
                   false, af0, false, bf, (short)0, acc0, false, false);
        acc1 = __builtin_amdgcn_wmma_f32_16x16x32_f16(
                   false, af1, false, bf, (short)0, acc1, false, false);
    }

    if (nok) {
#pragma unroll
        for (int g = 0; g < 8; ++g) {
            float v0 = acc0[g], v1 = acc1[g];
            if (relu) { v0 = fmaxf(v0, 0.f); v1 = fmaxf(v1, 0.f); }
            C[(size_t)(m0 + half * 8 + g) * N + n] = v0;
            C[(size_t)(m0 + 16 + half * 8 + g) * N + n] = v1;
        }
    }
}

// ---------------------------------------------------------------------------
// Support kernels (irregular parts: scatter, argmin, chamfer, elementwise)
// ---------------------------------------------------------------------------
__global__ void k_fill(float* p, float v, size_t n)
{
    size_t i = (size_t)blockIdx.x * blockDim.x + threadIdx.x;
    if (i < n) p[i] = v;
}

// conv weight pre-transpose: wt[t][ci][co] = w[co][ci][t]   (2x2x2 taps t)
__global__ void k_convw_t(const float* __restrict__ w, float* __restrict__ wt)
{
    size_t e = (size_t)blockIdx.x * blockDim.x + threadIdx.x;
    if (e >= (size_t)8 * HID * HID) return;
    int t  = (int)(e >> 16);
    int ci = (int)((e >> 8) & 255);
    int co = (int)(e & 255);
    wt[e] = w[(size_t)co * 2048 + ci * 8 + t];
}

// per-batch min/max of feature channels 0..2 (used as voxel coordinates)
__global__ void k_minmax3(const float* __restrict__ f, float* __restrict__ mnmx)
{
    __shared__ float smn[3][256], smx[3][256];
    int b = blockIdx.x, t = threadIdx.x;
    float mn[3] = {3.4e38f, 3.4e38f, 3.4e38f};
    float mx[3] = {-3.4e38f, -3.4e38f, -3.4e38f};
    for (int n = t; n < NPTS; n += 256) {
        const float* p = f + ((size_t)b * NPTS + n) * HID;
#pragma unroll
        for (int c = 0; c < 3; ++c) {
            float v = p[c];
            mn[c] = fminf(mn[c], v);
            mx[c] = fmaxf(mx[c], v);
        }
    }
#pragma unroll
    for (int c = 0; c < 3; ++c) { smn[c][t] = mn[c]; smx[c][t] = mx[c]; }
    __syncthreads();
    for (int s = 128; s > 0; s >>= 1) {
        if (t < s)
#pragma unroll
            for (int c = 0; c < 3; ++c) {
                smn[c][t] = fminf(smn[c][t], smn[c][t + s]);
                smx[c][t] = fmaxf(smx[c][t], smx[c][t + s]);
            }
        __syncthreads();
    }
    if (t < 3) { mnmx[b * 6 + t] = smn[t][0]; mnmx[b * 6 + 3 + t] = smx[t][0]; }
}

// scatter-add each point's 256-d feature into its voxel cell (one block/point)
__global__ void k_scatter(const float* __restrict__ f, const float* __restrict__ mnmx,
                          float* __restrict__ vox)
{
    int m = blockIdx.x, t = threadIdx.x;
    int b = m >> 11, n = m & (NPTS - 1);
    const float* p = f + ((size_t)b * NPTS + n) * HID;
    int cell = 0;
#pragma unroll
    for (int c = 0; c < 3; ++c) {
        float mn = mnmx[b * 6 + c], mx = mnmx[b * 6 + 3 + c];
        float v = (p[c] - mn) / (mx - mn) * (float)(RES - 1);
        int q = (int)v;                            // trunc-toward-zero like astype(int32)
        q = q < 0 ? 0 : (q > RES - 1 ? RES - 1 : q);
        cell = cell * RES + q;
    }
    atomicAdd(&vox[((size_t)b * NVOX + cell) * HID + t], p[t]);
}

// gather one 2x2x2 conv tap into a dense [B*216, 256] matrix (then WMMA GEMM)
__global__ void k_gather_tap(const float* __restrict__ vox, float* __restrict__ At, int t)
{
    size_t e = (size_t)blockIdx.x * blockDim.x + threadIdx.x;
    if (e >= (size_t)BATCH * OSP * HID) return;
    int c = (int)(e & (HID - 1));
    int m = (int)(e >> 8);
    int b = m / OSP, r = m - b * OSP;
    int od = r / 36, oh = (r / 6) % 6, ow = r % 6;
    int kd = (t >> 2) & 1, kh = (t >> 1) & 1, kw = t & 1;
    int cell = (2 * od + kd) * 144 + (2 * oh + kh) * 12 + (2 * ow + kw);
    At[e] = vox[((size_t)b * NVOX + cell) * HID + c];
}

__global__ void k_sqnorm(const float* __restrict__ cb, float* __restrict__ c2)
{
    int i = blockIdx.x * blockDim.x + threadIdx.x;
    if (i >= KC) return;
    float s = 0.f;
    for (int c = 0; c < HID; ++c) { float v = cb[(size_t)i * HID + c]; s += v * v; }
    c2[i] = s;
}

// VQ: one block per point. dist(n) = x2 + c2[n] - 2*dot[m,n]; argmin (first-min
// tie-break, matching jnp.argmin); gather codebook row; accumulate 1.25*mean sq.
__global__ void k_vq(const float* __restrict__ fk, int P, int strideRows,
                     const float* __restrict__ dot, const float* __restrict__ c2,
                     const float* __restrict__ cb, float* __restrict__ q,
                     float* __restrict__ scal, float lossScale)
{
    __shared__ float sx[256];
    __shared__ float sv[256];
    __shared__ int   si[256];
    int m = blockIdx.x, t = threadIdx.x;
    int b = m / P, r = m - b * P;
    const float* xr = fk + ((size_t)b * strideRows + r) * HID;
    float xv = xr[t];
    sx[t] = xv * xv;
    __syncthreads();
    for (int s = 128; s > 0; s >>= 1) { if (t < s) sx[t] += sx[t + s]; __syncthreads(); }
    float x2 = sx[0];

    const float* dr = dot + (size_t)m * KC;
    float best = 3.4e38f; int bi = KC;
    for (int n = t; n < KC; n += 256) {
        float d = c2[n] - 2.f * dr[n];
        if (d < best) { best = d; bi = n; }
    }
    sv[t] = best; si[t] = bi;
    __syncthreads();
    for (int s = 128; s > 0; s >>= 1) {
        if (t < s) {
            float v2 = sv[t + s]; int i2 = si[t + s];
            if (v2 < sv[t] || (v2 == sv[t] && i2 < si[t])) { sv[t] = v2; si[t] = i2; }
        }
        __syncthreads();
    }
    int bidx = si[0]; float bd = sv[0];
    q[(size_t)m * HID + t] = cb[(size_t)bidx * HID + t];
    if (t == 0) atomicAdd(&scal[0], lossScale * (x2 + bd));
}

__global__ void k_sub(float* __restrict__ a, const float* __restrict__ b, size_t n)
{
    size_t i = (size_t)blockIdx.x * blockDim.x + threadIdx.x;
    if (i < n) a[i] -= b[i];
}

// dst[b, r<rows, :] = a[b*aStride + r, :] + c[b*cStride + r, :]   (per-batch truncation)
__global__ void k_add2(float* __restrict__ dst, const float* __restrict__ a, int aStride,
                       const float* __restrict__ c, int cStride, int rows)
{
    size_t e = (size_t)blockIdx.x * blockDim.x + threadIdx.x;
    if (e >= (size_t)BATCH * rows * HID) return;
    int ch = (int)(e & (HID - 1));
    int m = (int)(e >> 8);
    int b = m / rows, r = m - b * rows;
    dst[((size_t)b * rows + r) * HID + ch] =
        a[((size_t)b * aStride + r) * HID + ch] + c[((size_t)b * cStride + r) * HID + ch];
}

__global__ void k_cham_row(const float* __restrict__ x, const float* __restrict__ y,
                           float* __restrict__ sum1)
{
    int i = blockIdx.x * blockDim.x + threadIdx.x;
    if (i >= BATCH * NPTS) return;
    int b = i >> 11;
    const float* px = x + (size_t)i * 3;
    const float* py = y + (size_t)b * OSP * 3;
    float best = 3.4e38f;
    for (int j = 0; j < OSP; ++j) {
        float d0 = px[0] - py[j * 3 + 0];
        float d1 = px[1] - py[j * 3 + 1];
        float d2 = px[2] - py[j * 3 + 2];
        float d = d0 * d0 + d1 * d1 + d2 * d2;
        best = fminf(best, d);
    }
    atomicAdd(sum1, best);
}

__global__ void k_cham_col(const float* __restrict__ x, const float* __restrict__ y,
                           float* __restrict__ sum2)
{
    int i = blockIdx.x * blockDim.x + threadIdx.x;
    if (i >= BATCH * OSP) return;
    int b = i / OSP;
    const float* py = y + (size_t)i * 3;
    const float* px = x + (size_t)b * NPTS * 3;
    float best = 3.4e38f;
    for (int j = 0; j < NPTS; ++j) {
        float d0 = py[0] - px[j * 3 + 0];
        float d1 = py[1] - px[j * 3 + 1];
        float d2 = py[2] - px[j * 3 + 2];
        float d = d0 * d0 + d1 * d1 + d2 * d2;
        best = fminf(best, d);
    }
    atomicAdd(sum2, best);
}

__global__ void k_final(const float* __restrict__ scal, float* __restrict__ out)
{
    if (blockIdx.x == 0 && threadIdx.x == 0) {
        float ch = scal[1] / (float)(BATCH * NPTS) + scal[2] / (float)(BATCH * OSP);
        float vq = scal[0];
        out[RECN + 0] = ch + vq;
        out[RECN + 1] = ch;
        out[RECN + 2] = vq;
    }
}

// ---------------------------------------------------------------------------
// Host-side orchestration
// ---------------------------------------------------------------------------
static inline void launch_gemm(hipStream_t s, const float* A, int rowsUsed, int strideRows,
                               const float* Wm, long long ldbK, long long ldbN, long long boff,
                               const float* bias, float* C, int K, int N, int accC, int relu)
{
    int M = rowsUsed * BATCH;                     // always a multiple of 32
    bool colB = (ldbK == 1), rowB = (ldbN == 1);
    if ((K % 32 == 0) && (N % 16 == 0) && (rowB || colB)) {
        int tilesM = M / 32, tilesN = N / 16;
        int blocks = (tilesM * tilesN + 7) / 8;
        if (colB)
            k_gemm_fast<1><<<blocks, 256, 0, s>>>(A, rowsUsed, strideRows, Wm, ldbK, ldbN,
                                                  boff, bias, C, M, K, N, tilesM, tilesN,
                                                  accC, relu);
        else
            k_gemm_fast<0><<<blocks, 256, 0, s>>>(A, rowsUsed, strideRows, Wm, ldbK, ldbN,
                                                  boff, bias, C, M, K, N, tilesM, tilesN,
                                                  accC, relu);
    } else {                                      // ragged K and/or N, row-major B
        int tilesM = M / 32, tilesN = (N + 15) / 16;
        int blocks = (tilesM * tilesN + 7) / 8;
        k_gemm_edge<<<blocks, 256, 0, s>>>(A, rowsUsed, strideRows, Wm, ldbK, boff,
                                           bias, C, M, K, N, tilesM, tilesN, relu);
    }
}

static inline void launch_mlp2(hipStream_t s, const float* A, int rowsUsed, int strideRows,
                               const float* w0, const float* b0,
                               const float* w1, const float* b1, int Nout,
                               float* mid, float* out)
{
    launch_gemm(s, A, rowsUsed, strideRows, w0, HID, 1, 0, b0, mid, HID, HID, 0, 1);
    launch_gemm(s, mid, rowsUsed, rowsUsed, w1, Nout, 1, 0, b1, out, HID, Nout, 0, 0);
}

struct UpsSet { const float *w0, *b0, *w1, *b1; };

// Factor-chain upsample with backward row-requirement propagation so we never
// compute rows the final [:, :target] truncation throws away.
static float* upsample_chain(hipStream_t s, const float* src, int nIn, int srcStride,
                             int target, const UpsSet& u4, const UpsSet& u8,
                             float* work, float* tmid, int* rowsOut)
{
    int fs[6]; int nst = 0; int n = nIn;
    while (n < target && nst < 6) {
        int q = target / n;
        int d4 = q > 4 ? q - 4 : 4 - q;
        int d8 = q > 8 ? q - 8 : 8 - q;
        int f = (d4 <= d8) ? 4 : 8;   // ties -> 4 (FACTORS order)
        fs[nst++] = f; n *= f;
    }
    if (nst == 0) { *rowsOut = nIn; return (float*)src; }
    int req[7];
    req[nst] = (n > target) ? target : n;
    for (int i = nst - 1; i >= 0; --i) req[i] = (req[i + 1] + fs[i] - 1) / fs[i];
    if (req[0] > nIn) req[0] = nIn;

    const float* cur = src; int curStride = srcStride;
    for (int i = 0; i < nst; ++i) {
        const UpsSet& u = (fs[i] == 4) ? u4 : u8;
        int ru = req[i];
        int Nout = HID * fs[i];
        launch_gemm(s, cur, ru, curStride, u.w0, HID, 1, 0, u.b0, tmid, HID, HID, 0, 1);
        launch_gemm(s, tmid, ru, ru, u.w1, Nout, 1, 0, u.b1, work, HID, Nout, 0, 0);
        cur = work; curStride = ru * fs[i];   // [B,ru,HID*f] == [B,ru*f,HID] row-major
    }
    *rowsOut = curStride;
    return work;
}

extern "C" void kernel_launch(void* const* d_in, const int* in_sizes, int n_in,
                              void* d_out, int out_size, void* d_ws, size_t ws_size,
                              hipStream_t stream)
{
    (void)in_sizes; (void)n_in; (void)out_size; (void)ws_size;

    // ---- input leaves (setup_inputs insertion order; (w,b) pairs in order) ----
    const float* x = (const float*)d_in[0];
    const float* encw[3] = {(const float*)d_in[1], (const float*)d_in[3], (const float*)d_in[5]};
    const float* encb[3] = {(const float*)d_in[2], (const float*)d_in[4], (const float*)d_in[6]};
    // dec_layers: 4 * (w0,b0,w1,b1) at 7..22 ; dec_final 23..26 ; phi 27..46
    const float* dfw0 = (const float*)d_in[23]; const float* dfb0 = (const float*)d_in[24];
    const float* dfw1 = (const float*)d_in[25]; const float* dfb1 = (const float*)d_in[26];
    const float* cb    = (const float*)d_in[47];
    const float* convw = (const float*)d_in[48];
    const float* convb = (const float*)d_in[49];
    UpsSet u4 = {(const float*)d_in[50], (const float*)d_in[51],
                 (const float*)d_in[52], (const float*)d_in[53]};
    UpsSet u8 = {(const float*)d_in[54], (const float*)d_in[55],
                 (const float*)d_in[56], (const float*)d_in[57]};
    auto PHI = [&](int k, int j) { return (const float*)d_in[27 + k * 4 + j]; };
    auto DL  = [&](int l, int j) { return (const float*)d_in[7 + l * 4 + j]; };

    // ---- workspace layout (~254 MB of f32) ----
    float* W = (float*)d_ws;
    size_t off = 0;
    auto alloc = [&](size_t nf) { float* p = W + off; off += nf; return p; };
    const size_t FB = (size_t)BATCH * NPTS * HID;      // 16.78M floats
    float* fbuf = alloc(FB);                           // residual features
    float* bufA = alloc(FB);                           // big ping (also vox / pz)
    float* bufB = alloc(FB);                           // big pong (also dot / mids)
    float* fk   = alloc((size_t)BATCH * OSP * HID);    // conv output [B,216,256]
    float* At   = alloc((size_t)BATCH * OSP * HID);    // conv tap gather
    float* Wt   = alloc((size_t)8 * HID * HID);        // transposed conv weights
    const int Pk[5] = {1, 4, 32, OSP, OSP};
    float* qk[5];
    for (int k = 0; k < 5; ++k) qk[k] = alloc((size_t)BATCH * Pk[k] * HID);
    float* fhat = alloc((size_t)BATCH * OSP * HID);
    float* hbuf = alloc((size_t)BATCH * OSP * HID);
    float* prb  = alloc((size_t)BATCH * OSP * HID);
    float* mnmx = alloc(BATCH * 6);
    float* c2   = alloc(KC);
    float* scal = alloc(8);                            // [vq, cham_row, cham_col]

    float* rec = (float*)d_out;

    // ---- init accumulators + codebook norms + transposed conv weights ----
    k_fill<<<1, 32, 0, stream>>>(scal, 0.f, 8);
    k_sqnorm<<<(KC + 255) / 256, 256, 0, stream>>>(cb, c2);
    k_convw_t<<<(8 * HID * HID) / 256, 256, 0, stream>>>(convw, Wt);

    // ---- encoder: 3 -> 256 -> 256 -> 256 ----
    launch_gemm(stream, x, NPTS, NPTS, encw[0], HID, 1, 0, encb[0], bufA, 3,   HID, 0, 1);
    launch_gemm(stream, bufA, NPTS, NPTS, encw[1], HID, 1, 0, encb[1], bufB, HID, HID, 0, 1);
    launch_gemm(stream, bufB, NPTS, NPTS, encw[2], HID, 1, 0, encb[2], fbuf, HID, HID, 0, 0);

    // ---- residual-VQ encode over 5 scales ----
    const size_t VOXN = (size_t)BATCH * NVOX * HID;
    for (int k = 0; k < 5; ++k) {
        // downsample: voxelize (scatter-add) then 2x2x2/s2 conv as 8 tap-GEMMs
        k_minmax3<<<BATCH, 256, 0, stream>>>(fbuf, mnmx);
        k_fill<<<(unsigned)((VOXN + 255) / 256), 256, 0, stream>>>(bufA, 0.f, VOXN);
        k_scatter<<<BATCH * NPTS, 256, 0, stream>>>(fbuf, mnmx, bufA);
        for (int t = 0; t < 8; ++t) {
            k_gather_tap<<<(BATCH * OSP * HID) / 256, 256, 0, stream>>>(bufA, At, t);
            // W_t(k=ci, n=co) = Wt[t*65536 + ci*256 + co]  (row-major fast path)
            launch_gemm(stream, At, OSP, OSP, Wt, HID, 1, (long long)t * HID * HID,
                        (t == 0) ? convb : nullptr, fk, HID, HID, (t == 0) ? 0 : 1, 0);
        }
        // VQ: dot = fk . cb^T (WMMA, col-major B fast path), then fused argmin
        int P = Pk[k], M = BATCH * P;
        launch_gemm(stream, fk, P, OSP, cb, 1, HID, 0, nullptr, bufB, HID, KC, 0, 0);
        float lossScale = 1.25f / ((float)M * (float)HID);
        k_vq<<<M, 256, 0, stream>>>(fk, P, OSP, bufB, c2, cb, qk[k], scal, lossScale);

        if (k < 4) {
            int zr = 0;
            float* z = upsample_chain(stream, qk[k], P, P, NPTS, u4, u8, bufA, bufB, &zr);
            // pz = mlp2(z, phi[k]) ; f -= pz
            launch_gemm(stream, z, NPTS, zr, PHI(k, 0), HID, 1, 0, PHI(k, 1), bufB, HID, HID, 0, 1);
            launch_gemm(stream, bufB, NPTS, NPTS, PHI(k, 2), HID, 1, 0, PHI(k, 3), bufA, HID, HID, 0, 0);
            k_sub<<<(unsigned)((FB + 255) / 256), 256, 0, stream>>>(fbuf, bufA, FB);
        }
    }

    // ---- decode ----
    // k=0: f_hat = mlp2(q0, phi[4])  (z is zeros)
    launch_mlp2(stream, qk[0], 1, 1, PHI(4, 0), PHI(4, 1), PHI(4, 2), PHI(4, 3), HID, bufB, fhat);
    int nf = 1;
    const int tgt[5] = {1, 4, 32, 256, 2048};
    for (int k = 1; k < 5; ++k) {
        int zr = 0;
        float* z = upsample_chain(stream, fhat, nf, nf, tgt[k], u4, u8, bufA, bufB, &zr);
        int P = Pk[k];
        launch_mlp2(stream, qk[k], P, P, PHI(k - 1, 0), PHI(k - 1, 1),
                    PHI(k - 1, 2), PHI(k - 1, 3), HID, bufB, prb);
        int m = (tgt[k] < P) ? tgt[k] : P;          // min(z rows, pr rows)
        k_add2<<<(BATCH * m * HID + 255) / 256, 256, 0, stream>>>(fhat, z, zr, prb, P, m);
        nf = m;
    }
    // decoder MLP stack on [B,216,256], final 256->3 straight into d_out
    float* srcp = fhat; float* dstp = hbuf;
    for (int l = 0; l < 4; ++l) {
        launch_mlp2(stream, srcp, OSP, OSP, DL(l, 0), DL(l, 1), DL(l, 2), DL(l, 3), HID, bufB, dstp);
        float* t = srcp; srcp = dstp; dstp = t;
    }
    launch_gemm(stream, srcp, OSP, OSP, dfw0, HID, 1, 0, dfb0, bufB, HID, HID, 0, 1);
    launch_gemm(stream, bufB, OSP, OSP, dfw1, 3, 1, 0, dfb1, rec, HID, 3, 0, 0);

    // ---- chamfer + scalar assembly ----
    k_cham_row<<<(BATCH * NPTS + 255) / 256, 256, 0, stream>>>(x, rec, scal + 1);
    k_cham_col<<<(BATCH * OSP + 255) / 256, 256, 0, stream>>>(x, rec, scal + 2);
    k_final<<<1, 32, 0, stream>>>(scal, rec);
}
